// Attention_32512902431074
// MI455X (gfx1250) — matrix operand order repaired and checked by reading, not compiled, not run on hardware
//
#include <hip/hip_runtime.h>

// ---------------------------------------------------------------------------
// CDNA5 (gfx1250) fused causal multi-head attention, f16 WMMA / f32 accum.
//   K1: QKV projection GEMM (128x128x64 tiles, 16 WMMA/wave/iter)
//   K2: flash attention, 1 wave per 16 query rows, online softmax
//   K3: output projection GEMM, A tile staged by the Tensor Data Mover
// ---------------------------------------------------------------------------

typedef _Float16 v16h __attribute__((ext_vector_type(16)));
typedef _Float16 v8h  __attribute__((ext_vector_type(8)));
typedef float    v8f  __attribute__((ext_vector_type(8)));
typedef unsigned int v4u __attribute__((ext_vector_type(4)));
typedef int      v4i  __attribute__((ext_vector_type(4)));
typedef int      v8i  __attribute__((ext_vector_type(8)));

union V16 { v16h v; v8h h[2]; };

#define B_   2
#define T_   2048
#define D_   1024
#define H_   16
#define HD_  64
#define SCALE_ 0.03125f   // D^-0.5 = 1/32

static __device__ inline v8f wmma_f16(v16h a, v16h b, v8f c) {
    return __builtin_amdgcn_wmma_f32_16x16x32_f16(
        false, a, false, b, (short)0, c, false, false);
}

// ---------------------------------------------------------------------------
// Kernel 1: QKV projection GEMM.  C[M=B*T, N=D] = x @ w + bias
// Block: 256 thr = 8 waves (4M x 2N), tile 128x128, BK=64; wave tile 32x64.
// ---------------------------------------------------------------------------
__global__ __launch_bounds__(256)
void qkv_gemm_kernel(const float* __restrict__ x,
                     const float* __restrict__ wq, const float* __restrict__ bq,
                     const float* __restrict__ wk, const float* __restrict__ bk,
                     const float* __restrict__ wv, const float* __restrict__ bv,
                     _Float16* __restrict__ Qh, _Float16* __restrict__ Kh,
                     _Float16* __restrict__ Vt)
{
    const int z   = blockIdx.z;
    const float* w   = (z == 0) ? wq : (z == 1) ? wk : wv;
    const float* bia = (z == 0) ? bq : (z == 1) ? bk : bv;

    const int n0   = blockIdx.x * 128;
    const int m0   = blockIdx.y * 128;
    const int tid  = threadIdx.x;
    const int wave = tid >> 5, lane = tid & 31;
    const int wm   = wave & 3, wn = wave >> 2;
    const int l15  = lane & 15, kh = lane >> 4;

    __shared__ _Float16 As[128 * 64];   // [m][k]
    __shared__ _Float16 Bs[128 * 64];   // [n][k] (transposed)

    v8f c[2][4] = {};

    for (int k0 = 0; k0 < D_; k0 += 64) {
        {   // stage A: 128x64 f32 -> f16, 32 contiguous k per thread
            int row = tid >> 1;
            int ks  = (tid & 1) * 32;
            const float* src = x + (size_t)(m0 + row) * D_ + k0 + ks;
            _Float16* dst = As + row * 64 + ks;
            #pragma unroll
            for (int c8 = 0; c8 < 4; ++c8) {
                v8h t;
                #pragma unroll
                for (int j = 0; j < 8; ++j) t[j] = (_Float16)src[c8 * 8 + j];
                *(v8h*)(dst + c8 * 8) = t;
            }
        }
        {   // stage B column-wise: w[k][n] -> Bs[n][k], coalesced across lanes
            int n  = tid & 127;
            int ks = (tid >> 7) * 32;
            const float* src = w + (size_t)(k0 + ks) * D_ + n0 + n;
            _Float16* dst = Bs + n * 64 + ks;
            #pragma unroll
            for (int c8 = 0; c8 < 4; ++c8) {
                v8h t;
                #pragma unroll
                for (int j = 0; j < 8; ++j)
                    t[j] = (_Float16)src[(size_t)(c8 * 8 + j) * D_];
                *(v8h*)(dst + c8 * 8) = t;
            }
            if (k0 + 64 < D_) __builtin_prefetch(src + (size_t)64 * D_, 0, 0);
        }
        __syncthreads();

        #pragma unroll
        for (int ks = 0; ks < 64; ks += 32) {
            V16 a[2];
            #pragma unroll
            for (int mi = 0; mi < 2; ++mi) {
                const _Float16* arow = As + (wm * 32 + mi * 16 + l15) * 64 + ks;
                a[mi].h[0] = *(const v8h*)(arow + kh * 8);
                a[mi].h[1] = *(const v8h*)(arow + 16 + kh * 8);
            }
            #pragma unroll
            for (int ni = 0; ni < 4; ++ni) {
                V16 b;
                const _Float16* bp = Bs + (wn * 64 + ni * 16 + l15) * 64 + ks + kh * 8;
                b.h[0] = *(const v8h*)(bp);
                b.h[1] = *(const v8h*)(bp + 16);
                c[0][ni] = wmma_f16(a[0].v, b.v, c[0][ni]);
                c[1][ni] = wmma_f16(a[1].v, b.v, c[1][ni]);
            }
        }
        __syncthreads();
    }

    // epilogue: bias, convert, scatter into head layouts
    #pragma unroll
    for (int ni = 0; ni < 4; ++ni) {
        const int col = n0 + wn * 64 + ni * 16 + l15;
        const int hh  = col >> 6, hd = col & 63;
        const float bb = bia[col];
        #pragma unroll
        for (int mi = 0; mi < 2; ++mi) {
            #pragma unroll
            for (int i = 0; i < 8; ++i) {
                const int mrow = m0 + wm * 32 + mi * 16 + i + 8 * kh;
                const int b = mrow >> 11, t = mrow & (T_ - 1);
                const float val = c[mi][ni][i] + bb;
                if (z == 0)
                    Qh[(size_t)((b * H_ + hh) * T_ + t) * HD_ + hd] = (_Float16)val;
                else if (z == 1)
                    Kh[(size_t)((b * H_ + hh) * T_ + t) * HD_ + hd] = (_Float16)val;
                else
                    Vt[(size_t)((b * H_ + hh) * HD_ + hd) * T_ + t] = (_Float16)val;
            }
        }
    }
}

// ---------------------------------------------------------------------------
// Kernel 2: causal flash attention.
// ---------------------------------------------------------------------------
__global__ __launch_bounds__(128)
void attn_kernel(const _Float16* __restrict__ Qh, const _Float16* __restrict__ Kh,
                 const _Float16* __restrict__ Vt, _Float16* __restrict__ attn)
{
    const int tid  = threadIdx.x;
    const int wave = tid >> 5, lane = tid & 31;
    const int l15  = lane & 15, kh = lane >> 4;
    const int bh   = blockIdx.y;
    const int b    = bh >> 4, h = bh & 15;
    const int q_base = (blockIdx.x * 4 + wave) * 16;

    const _Float16* Qp = Qh + (size_t)bh * T_ * HD_;
    const _Float16* Kp = Kh + (size_t)bh * T_ * HD_;
    const _Float16* Vp = Vt + (size_t)bh * HD_ * T_;

    __shared__ _Float16 lds_p[4][16 * 32];
    _Float16* pbuf = lds_p[wave];

    V16 aq0, aq1;
    {
        const _Float16* qrow = Qp + (size_t)(q_base + l15) * HD_;
        aq0.h[0] = *(const v8h*)(qrow + kh * 8);
        aq0.h[1] = *(const v8h*)(qrow + 16 + kh * 8);
        aq1.h[0] = *(const v8h*)(qrow + 32 + kh * 8);
        aq1.h[1] = *(const v8h*)(qrow + 48 + kh * 8);
    }

    v8f o0 = {}, o1 = {}, o2 = {}, o3 = {};
    float m_i[8], l_i[8];
    #pragma unroll
    for (int i = 0; i < 8; ++i) { m_i[i] = -1e30f; l_i[i] = 0.0f; }

    const int q_last = q_base + 15;
    for (int kb = 0; kb <= q_last; kb += 32) {
        v8f s0 = {}, s1 = {};
        #pragma unroll
        for (int f = 0; f < 2; ++f) {
            V16 bk0, bk1;
            const _Float16* krow = Kp + (size_t)(kb + f * 16 + l15) * HD_;
            bk0.h[0] = *(const v8h*)(krow + kh * 8);
            bk0.h[1] = *(const v8h*)(krow + 16 + kh * 8);
            bk1.h[0] = *(const v8h*)(krow + 32 + kh * 8);
            bk1.h[1] = *(const v8h*)(krow + 48 + kh * 8);
            v8f s = {};
            s = wmma_f16(aq0.v, bk0.v, s);
            s = wmma_f16(aq1.v, bk1.v, s);
            if (f) s1 = s; else s0 = s;
        }

        #pragma unroll
        for (int i = 0; i < 8; ++i) {
            const int q    = q_base + i + 8 * kh;
            const int key0 = kb + l15;
            const int key1 = kb + 16 + l15;
            float v0 = s0[i] * SCALE_;
            float v1 = s1[i] * SCALE_;
            if (key0 > q) v0 = -1e30f;
            if (key1 > q) v1 = -1e30f;
            float mx = fmaxf(v0, v1);
            #pragma unroll
            for (int off = 1; off < 16; off <<= 1)
                mx = fmaxf(mx, __shfl_xor(mx, off, 32));
            const float mnew = fmaxf(m_i[i], mx);
            const float fac  = __expf(m_i[i] - mnew);
            m_i[i] = mnew;
            const float p0 = __expf(v0 - mnew);
            const float p1 = __expf(v1 - mnew);
            float ps = p0 + p1;
            #pragma unroll
            for (int off = 1; off < 16; off <<= 1)
                ps += __shfl_xor(ps, off, 32);
            l_i[i] = l_i[i] * fac + ps;
            o0[i] *= fac; o1[i] *= fac; o2[i] *= fac; o3[i] *= fac;
            pbuf[(i + 8 * kh) * 32 + l15]      = (_Float16)p0;
            pbuf[(i + 8 * kh) * 32 + 16 + l15] = (_Float16)p1;
        }
        asm volatile("s_wait_dscnt 0" ::: "memory");

        V16 ap;
        {
            const _Float16* prow = pbuf + l15 * 32;
            ap.h[0] = *(const v8h*)(prow + kh * 8);
            ap.h[1] = *(const v8h*)(prow + 16 + kh * 8);
        }

        #pragma unroll
        for (int d = 0; d < 4; ++d) {
            V16 bv;
            const _Float16* vrow = Vp + (size_t)(d * 16 + l15) * T_ + kb + kh * 8;
            bv.h[0] = *(const v8h*)(vrow);
            bv.h[1] = *(const v8h*)(vrow + 16);
            if      (d == 0) o0 = wmma_f16(ap.v, bv.v, o0);
            else if (d == 1) o1 = wmma_f16(ap.v, bv.v, o1);
            else if (d == 2) o2 = wmma_f16(ap.v, bv.v, o2);
            else             o3 = wmma_f16(ap.v, bv.v, o3);
        }
    }

    #pragma unroll
    for (int i = 0; i < 8; ++i) {
        const int q = q_base + i + 8 * kh;
        const float inv = 1.0f / l_i[i];
        const size_t rowoff = ((size_t)(b * T_ + q)) * D_ + h * HD_;
        attn[rowoff +  0 + l15] = (_Float16)(o0[i] * inv);
        attn[rowoff + 16 + l15] = (_Float16)(o1[i] * inv);
        attn[rowoff + 32 + l15] = (_Float16)(o2[i] * inv);
        attn[rowoff + 48 + l15] = (_Float16)(o3[i] * inv);
    }
}

// ---------------------------------------------------------------------------
// Kernel 3: output projection. out[M, D] = attn(f16) @ wo + bo  (f32 out)
// A tile (128 x 64 f16) staged by the Tensor Data Mover into LDS.
// ---------------------------------------------------------------------------
__global__ __launch_bounds__(256)
void proj_gemm_kernel(const _Float16* __restrict__ attn,
                      const float* __restrict__ wo, const float* __restrict__ bo,
                      float* __restrict__ out)
{
    const int n0   = blockIdx.x * 128;
    const int m0   = blockIdx.y * 128;
    const int tid  = threadIdx.x;
    const int wave = tid >> 5, lane = tid & 31;
    const int wm   = wave & 3, wn = wave >> 2;
    const int l15  = lane & 15, kh = lane >> 4;

    __shared__ _Float16 As[128 * 64];   // [m][k]  (TDM destination)
    __shared__ _Float16 Bs[128 * 64];   // [n][k]

    v8f c[2][4] = {};

    for (int k0 = 0; k0 < D_; k0 += 64) {
        if (wave == 0) {
            // ---- Tensor DMA descriptor (D#): 2-D tile 64(k) x 128(m), f16 ----
            const unsigned long long ga =
                (unsigned long long)(uintptr_t)(attn + (size_t)m0 * D_ + k0);
            const unsigned int lds_addr = (unsigned int)(uintptr_t)As;
            v4u g0;
            g0[0] = 1u;                                    // count=1, user mode
            g0[1] = lds_addr;                              // lds_addr
            g0[2] = (unsigned int)ga;                      // global_addr[31:0]
            g0[3] = (unsigned int)((ga >> 32) & 0x01FFFFFFu) | (2u << 30); // [56:32] | type=2
            v8i g1;
            g1[0] = (int)(1u << 16);                       // data_size=2B, mask=0
            g1[1] = (int)((unsigned)(D_ & 0xFFFF) << 16);  // tensor_dim0[15:0]
            g1[2] = (int)(((unsigned)D_ >> 16) | ((unsigned)((B_*T_) & 0xFFFF) << 16));
            g1[3] = (int)(((unsigned)(B_*T_) >> 16) | (64u << 16));  // tile_dim0=64
            g1[4] = (int)128u;                             // tile_dim1=128, tile_dim2=0
            g1[5] = (int)(unsigned)D_;                     // tensor_dim0_stride=1024
            g1[6] = 0;
            g1[7] = 0;
            v4i g2 = {0, 0, 0, 0};
            v4i g3 = {0, 0, 0, 0};
            v8i g4 = {0, 0, 0, 0, 0, 0, 0, 0};
            __builtin_amdgcn_tensor_load_to_lds(g0, g1, g2, g3, g4, 0);
            __builtin_amdgcn_s_wait_tensorcnt(0);
        }
        {   // stage B column-wise: wo[k][n] -> Bs[n][k]
            int n  = tid & 127;
            int ks = (tid >> 7) * 32;
            const float* src = wo + (size_t)(k0 + ks) * D_ + n0 + n;
            _Float16* dst = Bs + n * 64 + ks;
            #pragma unroll
            for (int c8 = 0; c8 < 4; ++c8) {
                v8h t;
                #pragma unroll
                for (int j = 0; j < 8; ++j)
                    t[j] = (_Float16)src[(size_t)(c8 * 8 + j) * D_];
                *(v8h*)(dst + c8 * 8) = t;
            }
            if (k0 + 64 < D_) __builtin_prefetch(src + (size_t)64 * D_, 0, 0);
        }
        __syncthreads();

        #pragma unroll
        for (int ks = 0; ks < 64; ks += 32) {
            V16 a[2];
            #pragma unroll
            for (int mi = 0; mi < 2; ++mi) {
                const _Float16* arow = As + (wm * 32 + mi * 16 + l15) * 64 + ks;
                a[mi].h[0] = *(const v8h*)(arow + kh * 8);
                a[mi].h[1] = *(const v8h*)(arow + 16 + kh * 8);
            }
            #pragma unroll
            for (int ni = 0; ni < 4; ++ni) {
                V16 b;
                const _Float16* bp = Bs + (wn * 64 + ni * 16 + l15) * 64 + ks + kh * 8;
                b.h[0] = *(const v8h*)(bp);
                b.h[1] = *(const v8h*)(bp + 16);
                c[0][ni] = wmma_f16(a[0].v, b.v, c[0][ni]);
                c[1][ni] = wmma_f16(a[1].v, b.v, c[1][ni]);
            }
        }
        __syncthreads();
    }

    #pragma unroll
    for (int ni = 0; ni < 4; ++ni) {
        const int col = n0 + wn * 64 + ni * 16 + l15;
        const float bb = bo[col];
        #pragma unroll
        for (int mi = 0; mi < 2; ++mi) {
            #pragma unroll
            for (int i = 0; i < 8; ++i) {
                const int mrow = m0 + wm * 32 + mi * 16 + i + 8 * kh;
                out[(size_t)mrow * D_ + col] = c[mi][ni][i] + bb;
            }
        }
    }
}

// ---------------------------------------------------------------------------
extern "C" void kernel_launch(void* const* d_in, const int* in_sizes, int n_in,
                              void* d_out, int out_size, void* d_ws, size_t ws_size,
                              hipStream_t stream) {
    const float* x  = (const float*)d_in[0];
    const float* wq = (const float*)d_in[1];
    const float* bq = (const float*)d_in[2];
    const float* wk = (const float*)d_in[3];
    const float* bk = (const float*)d_in[4];
    const float* wv = (const float*)d_in[5];
    const float* bv = (const float*)d_in[6];
    const float* wo = (const float*)d_in[7];
    const float* bo = (const float*)d_in[8];
    float* out = (float*)d_out;

    _Float16* ws = (_Float16*)d_ws;
    const size_t nQ = (size_t)B_ * H_ * T_ * HD_;   // 4Mi f16 each
    _Float16* Qh   = ws;
    _Float16* Kh   = ws + nQ;
    _Float16* Vt   = ws + 2 * nQ;
    _Float16* attn = ws + 3 * nQ;

    dim3 blk256(256);
    dim3 g1(D_ / 128, (B_ * T_) / 128, 3);
    hipLaunchKernelGGL(qkv_gemm_kernel, g1, blk256, 0, stream,
                       x, wq, bq, wk, bk, wv, bv, Qh, Kh, Vt);

    dim3 g2(T_ / 64, B_ * H_);
    hipLaunchKernelGGL(attn_kernel, g2, dim3(128), 0, stream, Qh, Kh, Vt, attn);

    dim3 g3(D_ / 128, (B_ * T_) / 128);
    hipLaunchKernelGGL(proj_gemm_kernel, g3, blk256, 0, stream, attn, wo, bo, out);
}
